// Decoder_32366873543325
// MI455X (gfx1250) — matrix operand order repaired and checked by reading, not compiled, not run on hardware
//
#include <hip/hip_runtime.h>

// ---------------------------------------------------------------------------
// Problem constants (match reference)
// ---------------------------------------------------------------------------
#define BB 8
#define STQ 512
#define SSK 576
#define TDIM 1024
#define SDIM 512
#define NLAYER 6
#define NH 16
#define HDD 64
#define FFD 4096
#define HD (NH * HDD)   // 1024

// GEMM tiling: block tile 128x64, K-stage 64 (two WMMA k-steps per barrier)
#define BM 128
#define BN 64
#define BK 64
#define LDSW 36         // LDS row stride in dwords: 144B (16B-aligned, conflict-free)

typedef __attribute__((ext_vector_type(16))) __bf16 bf16x16;
typedef __attribute__((ext_vector_type(8)))  float  f32x8;

// fp32 -> bf16 round-to-nearest-even
__device__ __forceinline__ unsigned short f2bf(float f) {
    union { float f; unsigned int u; } c; c.f = f;
    unsigned int u = c.u;
    return (unsigned short)((u + 0x7FFFu + ((u >> 16) & 1u)) >> 16);
}
__device__ __forceinline__ unsigned int pack_bf16(float a, float b) {
    return (unsigned int)f2bf(a) | ((unsigned int)f2bf(b) << 16);
}
__device__ __forceinline__ float bf2f(unsigned short h) {
    union { unsigned int u; float f; } c; c.u = ((unsigned int)h) << 16;
    return c.f;
}

// CDNA5 async global->LDS 16B copy (ASYNCcnt-tracked, no VGPR round-trip).
// LDS destination address = low 32 bits of the generic pointer to shared mem.
__device__ __forceinline__ void async_copy16(unsigned int lds_off, const void* g) {
    asm volatile("global_load_async_to_lds_b128 %0, %1, off"
                 :: "v"(lds_off), "v"(g)
                 : "memory");
}
__device__ __forceinline__ void wait_async0() {
#if __has_builtin(__builtin_amdgcn_s_wait_asynccnt)
    __builtin_amdgcn_s_wait_asynccnt(0);
#else
    asm volatile("s_wait_asynccnt 0x0" ::: "memory");
#endif
}

// ---------------------------------------------------------------------------
// Generic batched bf16 GEMM:  C = act( alpha * A @ B + bias + residual )
//   A: packed bf16 [M,K] row-major (lda elements, even) -> K-pairs contiguous
//   B: flags&1 == 0: source row n holds K-pairs of output column n contiguously
//      (pre-transposed weights Wt[N][K/2] or q@k^T). ldb = elements per row.
//      flags&1 == 1: bf16 [K,N] row-major (P@V): pack pairs while staging.
//   flags&2 = ReLU, flags&4 = bf16 output (else fp32 + optional residual)
//   batch z = b*H + h with per-operand element strides.
// Pipeline: double-buffered LDS, async copies for stage s+1 overlap WMMAs of s.
// ---------------------------------------------------------------------------
__global__ void __launch_bounds__(256)
gemm_bf16_kernel(const unsigned short* __restrict__ A,
                 const unsigned short* __restrict__ Bmat,
                 const float* __restrict__ bias, const float* __restrict__ Res,
                 void* __restrict__ Cv,
                 int M, int N, int K, int lda, int ldb, int ldc,
                 long long sAb, long long sAh, long long sBb, long long sBh,
                 long long sCb, long long sCh, int H, float alpha, int flags)
{
    const int bPK  = flags & 1;
    const int relu = flags & 2;
    const int obf  = flags & 4;

    const int z = blockIdx.z;
    const int b = z / H, h = z % H;
    const unsigned short* Ab = A    + (size_t)((long long)b * sAb + (long long)h * sAh);
    const unsigned short* Bb = Bmat + (size_t)((long long)b * sBb + (long long)h * sBh);
    const long long coff = (long long)b * sCb + (long long)h * sCh;

    const int m0 = blockIdx.y * BM;
    const int n0 = blockIdx.x * BN;

    __shared__ __align__(16) unsigned int As[2][BM][LDSW];
    __shared__ __align__(16) unsigned int Bs[2][BN][LDSW];

    const int tid  = threadIdx.x;
    const int lane = tid & 31;
    const int wave = tid >> 5;
    const int wm   = wave >> 1;     // 0..3 : 32-row group
    const int wn   = wave & 1;      // 0..1 : 32-col group
    const int half = lane >> 4;
    const int l15  = lane & 15;

    f32x8 acc00 = {0.f,0.f,0.f,0.f,0.f,0.f,0.f,0.f};
    f32x8 acc01 = acc00, acc10 = acc00, acc11 = acc00;

    const unsigned int* Adw = (const unsigned int*)Ab;   // bases are even elements
    const unsigned int* Bdw = (const unsigned int*)Bb;
    const int nB = tid & 63;
    const int gB = tid >> 6;

    // ---- staging helpers ----
    auto issueA = [&](int k0, int buf) {
#pragma unroll
        for (int i = 0; i < 4; ++i) {
            int p = i * 256 + tid;      // 1024 16B slots (128 rows x 32 dwords)
            int r = p >> 3, f = p & 7;
            const void* g = (const void*)(Adw + (((size_t)(m0 + r) * lda) >> 1) + (k0 >> 1) + 4 * f);
            async_copy16((unsigned int)(uintptr_t)&As[buf][r][4 * f], g);
        }
    };
    auto issueB0 = [&](int k0, int buf) {
#pragma unroll
        for (int i = 0; i < 2; ++i) {
            int p = i * 256 + tid;      // 512 16B slots (64 rows x 32 dwords)
            int r = p >> 3, f = p & 7;
            const void* g = (const void*)(Bdw + (((size_t)(n0 + r) * ldb) >> 1) + (k0 >> 1) + 4 * f);
            async_copy16((unsigned int)(uintptr_t)&Bs[buf][r][4 * f], g);
        }
    };
    unsigned int blo[8], bhi[8];        // P@V pack pipeline registers
    auto issueB1 = [&](int k0) {
#pragma unroll
        for (int i = 0; i < 8; ++i) {
            int kp = gB * 8 + i, k = k0 + 2 * kp;
            blo[i] = Bb[(size_t)k       * ldb + n0 + nB];
            bhi[i] = Bb[(size_t)(k + 1) * ldb + n0 + nB];
        }
    };
    auto finishB1 = [&](int buf) {
#pragma unroll
        for (int i = 0; i < 8; ++i) {
            int kp = gB * 8 + i;
            Bs[buf][nB][kp] = blo[i] | (bhi[i] << 16);
        }
    };

    // ---- pipelined main loop ----
    const int nk = K / BK;
    issueA(0, 0);
    if (!bPK) issueB0(0, 0); else issueB1(0);

    for (int s = 0; s < nk; ++s) {
        const int cur = s & 1;
        if (bPK) finishB1(cur);
        wait_async0();
        __syncthreads();
        if (s + 1 < nk) {
            const int k0n = (s + 1) * BK;
            issueA(k0n, cur ^ 1);
            if (!bPK) issueB0(k0n, cur ^ 1); else issueB1(k0n);
        }

        union U { unsigned int u[8]; bf16x16 v; };
        const int ra0 = wm * 32 + l15, ra1 = ra0 + 16;
        const int cb0 = wn * 32 + l15, cb1 = cb0 + 16;
#pragma unroll
        for (int st = 0; st < 2; ++st) {
            U a0, a1, bv0, bv1;
#pragma unroll
            for (int i = 0; i < 8; ++i) {
                int kpA = st * 16 + half * 8 + i;
                a0.u[i]  = As[cur][ra0][kpA];
                a1.u[i]  = As[cur][ra1][kpA];
                int kpB = st * 16 + half * 8 + i;
                bv0.u[i] = Bs[cur][cb0][kpB];
                bv1.u[i] = Bs[cur][cb1][kpB];
            }
            acc00 = __builtin_amdgcn_wmma_f32_16x16x32_bf16(false, a0.v, false, bv0.v,
                                                            (short)0, acc00, false, false);
            acc01 = __builtin_amdgcn_wmma_f32_16x16x32_bf16(false, a0.v, false, bv1.v,
                                                            (short)0, acc01, false, false);
            acc10 = __builtin_amdgcn_wmma_f32_16x16x32_bf16(false, a1.v, false, bv0.v,
                                                            (short)0, acc10, false, false);
            acc11 = __builtin_amdgcn_wmma_f32_16x16x32_bf16(false, a1.v, false, bv1.v,
                                                            (short)0, acc11, false, false);
        }
    }

    // ---- epilogue: row = base + half*8 + j, col = base + (lane&15) ----
    float*          Cf = (float*)Cv          + coff;
    unsigned short* Ch = (unsigned short*)Cv + coff;
    const float*    Rb = Res ? Res + coff : nullptr;
    auto store_tile = [&](const f32x8& acc, int ti, int tj) {
        int row0 = m0 + wm * 32 + ti * 16 + half * 8;
        int col  = n0 + wn * 32 + tj * 16 + l15;
        float bi = bias ? bias[col] : 0.f;
#pragma unroll
        for (int j = 0; j < 8; ++j) {
            int row = row0 + j;
            float v = acc[j] * alpha + bi;
            if (Rb)   v += Rb[(size_t)row * ldc + col];
            if (relu) v = fmaxf(v, 0.f);
            if (obf)  Ch[(size_t)row * ldc + col] = f2bf(v);
            else      Cf[(size_t)row * ldc + col] = v;
        }
    };
    store_tile(acc00, 0, 0);
    store_tile(acc01, 0, 1);
    store_tile(acc10, 1, 0);
    store_tile(acc11, 1, 1);
}

// ---------------------------------------------------------------------------
// Weight transpose+pack: fp32 W[K,N] -> Wt[N][K/2] dwords (bf16 K-pairs)
// ---------------------------------------------------------------------------
__global__ void __launch_bounds__(256)
wt_pack_kernel(const float* __restrict__ W, unsigned int* __restrict__ Wt, int K, int N)
{
    int gid = blockIdx.x * 256 + threadIdx.x;
    int total = (K >> 1) * N;
    if (gid >= total) return;
    int kp = gid / N;                    // n fast -> coalesced reads
    int n  = gid % N;
    float lo = W[(size_t)(2 * kp)     * N + n];
    float hi = W[(size_t)(2 * kp + 1) * N + n];
    Wt[(size_t)n * (K >> 1) + kp] = pack_bf16(lo, hi);
}

// fp32 -> packed bf16, same layout (for src, converted once per call)
__global__ void __launch_bounds__(256)
cvt_bf16_kernel(const float* __restrict__ in, unsigned int* __restrict__ out, long long ndw)
{
    long long i = (long long)blockIdx.x * 256 + threadIdx.x;
    if (i < ndw) out[i] = pack_bf16(in[2 * i], in[2 * i + 1]);
}

// ---------------------------------------------------------------------------
// LayerNorm: fp32 in, packed bf16 out; one block per row
// ---------------------------------------------------------------------------
__global__ void __launch_bounds__(256)
ln_kernel(const float* __restrict__ x, const float* __restrict__ g,
          const float* __restrict__ bta, unsigned int* __restrict__ y, int dim)
{
    __shared__ float r1[256], r2[256];
    const int row = blockIdx.x;
    const float* xr = x + (size_t)row * dim;
    float a = 0.f, b = 0.f;
    for (int i = threadIdx.x; i < dim; i += 256) { float v = xr[i]; a += v; b += v * v; }
    r1[threadIdx.x] = a; r2[threadIdx.x] = b;
    __syncthreads();
    for (int off = 128; off > 0; off >>= 1) {
        if (threadIdx.x < off) { r1[threadIdx.x] += r1[threadIdx.x + off];
                                 r2[threadIdx.x] += r2[threadIdx.x + off]; }
        __syncthreads();
    }
    const float mean = r1[0] / dim;
    const float var  = r2[0] / dim - mean * mean;
    const float rstd = rsqrtf(var + 1e-5f);
    unsigned int* yr = y + (size_t)row * (dim >> 1);
    for (int i = threadIdx.x; i < (dim >> 1); i += 256) {
        float u0 = (xr[2 * i]     - mean) * rstd * g[2 * i]     + bta[2 * i];
        float u1 = (xr[2 * i + 1] - mean) * rstd * g[2 * i + 1] + bta[2 * i + 1];
        yr[i] = pack_bf16(u0, u1);
    }
}

// ---------------------------------------------------------------------------
// Masked softmax over bf16 scores [B,H,ST,Sk], in place; block per row
// ---------------------------------------------------------------------------
__global__ void __launch_bounds__(128)
softmax_kernel(unsigned short* __restrict__ scores, const int* __restrict__ mask,
               int Sk, int H, int ST, long long mb, long long mq)
{
    const int idx  = blockIdx.x;
    const int qrow = idx % ST;
    const int bh   = idx / ST;
    const int b    = bh / H;
    unsigned short* row = scores + (size_t)idx * Sk;
    const int* mrow = mask + (long long)b * mb + (long long)qrow * mq;

    __shared__ float buf[SSK];
    __shared__ float red[128];
    const int tid = threadIdx.x;

    float mx = -3.0e38f;
    for (int i = tid; i < Sk; i += 128) {
        float v = bf2f(row[i]);
        if (mrow[i] == 0) v = -1e9f;
        buf[i] = v;
        mx = fmaxf(mx, v);
    }
    red[tid] = mx; __syncthreads();
    for (int off = 64; off > 0; off >>= 1) {
        if (tid < off) red[tid] = fmaxf(red[tid], red[tid + off]);
        __syncthreads();
    }
    mx = red[0]; __syncthreads();

    float sum = 0.f;
    for (int i = tid; i < Sk; i += 128) { float e = __expf(buf[i] - mx); buf[i] = e; sum += e; }
    red[tid] = sum; __syncthreads();
    for (int off = 64; off > 0; off >>= 1) {
        if (tid < off) red[tid] += red[tid + off];
        __syncthreads();
    }
    const float inv = 1.0f / red[0];
    for (int i = tid; i < Sk; i += 128) row[i] = f2bf(buf[i] * inv);
}

// ---------------------------------------------------------------------------
// Host side
// ---------------------------------------------------------------------------
static inline void launch_gemm(hipStream_t s,
    const void* A, const void* Bm, const float* bias, const float* res, void* C,
    int M, int N, int K, int lda, int ldb, int ldc,
    long long sAb, long long sAh, long long sBb, long long sBh,
    long long sCb, long long sCh, int H, int batches, float alpha, int flags)
{
    dim3 grid(N / BN, M / BM, batches);
    gemm_bf16_kernel<<<grid, 256, 0, s>>>((const unsigned short*)A, (const unsigned short*)Bm,
                                          bias, res, C, M, N, K, lda, ldb, ldc,
                                          sAb, sAh, sBb, sBh, sCb, sCh, H, alpha, flags);
}

static inline void launch_wt(hipStream_t s, const float* W, void* Wt, int K, int N)
{
    int total = (K >> 1) * N;
    wt_pack_kernel<<<(total + 255) / 256, 256, 0, s>>>(W, (unsigned int*)Wt, K, N);
}

extern "C" void kernel_launch(void* const* d_in, const int* in_sizes, int n_in,
                              void* d_out, int out_size, void* d_ws, size_t ws_size,
                              hipStream_t stream)
{
    const float* src      = (const float*)d_in[0];
    const int*   tgt_mask = (const int*)  d_in[2];
    const int*   src_mask = (const int*)  d_in[3];
    const float* norm_g   = (const float*)d_in[4];
    const float* norm_b   = (const float*)d_in[5];
    const float* sa_kv    = (const float*)d_in[6];
    const float* sa_q     = (const float*)d_in[7];
    const float* sa_ow    = (const float*)d_in[8];
    const float* sa_ob    = (const float*)d_in[9];
    const float* ca_kv    = (const float*)d_in[10];
    const float* ca_q     = (const float*)d_in[11];
    const float* ca_ow    = (const float*)d_in[12];
    const float* ca_ob    = (const float*)d_in[13];
    const float* ff_w1    = (const float*)d_in[14];
    const float* ff_b1    = (const float*)d_in[15];
    const float* ff_w2    = (const float*)d_in[16];
    const float* ff_b2    = (const float*)d_in[17];
    const float* pre_g    = (const float*)d_in[18];
    const float* pre_b    = (const float*)d_in[19];

    float* t = (float*)d_out;                       // running activations [B,ST,TD] fp32
    const int Mq = BB * STQ;                        // 4096
    const int Ms = BB * SSK;                        // 4608
    const int BH = BB * NH;
    const float scale = 0.125f;                     // D^-0.5

    // ---- workspace carve-up ----
    char* w = (char*)d_ws;
    unsigned short* tn   = (unsigned short*)w;  w += (size_t)Mq * TDIM * 2;      // LN out
    unsigned short* qb   = (unsigned short*)w;  w += (size_t)Mq * HD * 2;        // q
    unsigned short* attn = (unsigned short*)w;  w += (size_t)Mq * HD * 2;        // attn ctx
    unsigned short* kv   = (unsigned short*)w;  w += (size_t)Ms * 2 * HD * 2;    // k|v
    unsigned short* srcb = (unsigned short*)w;  w += (size_t)Ms * SDIM * 2;      // src bf16
    unsigned short* wt   = (unsigned short*)w;  w += (size_t)FFD * TDIM * 2;     // Wt scratch
    unsigned short* sc   = (unsigned short*)w;                                   // scores / FF hidden

    // t = tgt ; srcb = bf16(src)
    hipMemcpyAsync(t, d_in[1], (size_t)Mq * TDIM * sizeof(float),
                   hipMemcpyDeviceToDevice, stream);
    {
        long long ndw = ((long long)Ms * SDIM) >> 1;
        cvt_bf16_kernel<<<(int)((ndw + 255) / 256), 256, 0, stream>>>(src, (unsigned int*)srcb, ndw);
    }

    for (int l = 0; l < NLAYER; ++l) {
        const float* skv = sa_kv + (size_t)l * TDIM * 2 * HD;
        const float* sq  = sa_q  + (size_t)l * TDIM * HD;
        const float* sow = sa_ow + (size_t)l * HD * TDIM;
        const float* sob = sa_ob + (size_t)l * TDIM;
        const float* ckv = ca_kv + (size_t)l * SDIM * 2 * HD;
        const float* cq  = ca_q  + (size_t)l * TDIM * HD;
        const float* cow = ca_ow + (size_t)l * HD * TDIM;
        const float* cob = ca_ob + (size_t)l * TDIM;
        const float* w1  = ff_w1 + (size_t)l * TDIM * FFD;
        const float* b1  = ff_b1 + (size_t)l * FFD;
        const float* w2  = ff_w2 + (size_t)l * FFD * TDIM;
        const float* b2  = ff_b2 + (size_t)l * TDIM;
        const float* pg  = pre_g + (size_t)l * TDIM;
        const float* pb  = pre_b + (size_t)l * TDIM;

        // ===================== self-attention =====================
        ln_kernel<<<Mq, 256, 0, stream>>>(t, norm_g, norm_b, (unsigned int*)tn, TDIM);
        launch_wt(stream, skv, wt, TDIM, 2 * HD);
        launch_gemm(stream, tn, wt, nullptr, nullptr, kv,
                    Mq, 2 * HD, TDIM, TDIM, TDIM, 2 * HD,
                    0,0,0,0,0,0, 1, 1, 1.f, 4);
        launch_wt(stream, sq, wt, TDIM, HD);
        launch_gemm(stream, tn, wt, nullptr, nullptr, qb,
                    Mq, HD, TDIM, TDIM, TDIM, HD,
                    0,0,0,0,0,0, 1, 1, 1.f, 4);
        // scores = scale * q @ k^T  (k rows contiguous in d -> packed-B path)
        launch_gemm(stream, qb, kv, nullptr, nullptr, sc,
                    STQ, STQ, HDD, HD, 2 * HD, STQ,
                    (long long)STQ * HD, HDD,
                    (long long)STQ * 2 * HD, HDD,
                    (long long)NH * STQ * STQ, (long long)STQ * STQ,
                    NH, BH, scale, 4);
        softmax_kernel<<<BH * STQ, 128, 0, stream>>>(sc, tgt_mask, STQ, NH, STQ,
                                                     (long long)STQ * STQ, (long long)STQ);
        // attn = P @ V (V row-major over K -> pack-staging path)
        launch_gemm(stream, sc, kv + HD, nullptr, nullptr, attn,
                    STQ, HDD, STQ, STQ, 2 * HD, HD,
                    (long long)NH * STQ * STQ, (long long)STQ * STQ,
                    (long long)STQ * 2 * HD, HDD,
                    (long long)STQ * HD, HDD,
                    NH, BH, 1.f, 1 | 4);
        launch_wt(stream, sow, wt, HD, TDIM);
        launch_gemm(stream, attn, wt, sob, t, t,
                    Mq, TDIM, HD, HD, HD, TDIM,
                    0,0,0,0,0,0, 1, 1, 1.f, 0);

        // ===================== cross-attention =====================
        ln_kernel<<<Mq, 256, 0, stream>>>(t, norm_g, norm_b, (unsigned int*)tn, TDIM);
        launch_wt(stream, ckv, wt, SDIM, 2 * HD);
        launch_gemm(stream, srcb, wt, nullptr, nullptr, kv,
                    Ms, 2 * HD, SDIM, SDIM, SDIM, 2 * HD,
                    0,0,0,0,0,0, 1, 1, 1.f, 4);
        launch_wt(stream, cq, wt, TDIM, HD);
        launch_gemm(stream, tn, wt, nullptr, nullptr, qb,
                    Mq, HD, TDIM, TDIM, TDIM, HD,
                    0,0,0,0,0,0, 1, 1, 1.f, 4);
        launch_gemm(stream, qb, kv, nullptr, nullptr, sc,
                    STQ, SSK, HDD, HD, 2 * HD, SSK,
                    (long long)STQ * HD, HDD,
                    (long long)SSK * 2 * HD, HDD,
                    (long long)NH * STQ * SSK, (long long)STQ * SSK,
                    NH, BH, scale, 4);
        softmax_kernel<<<BH * STQ, 128, 0, stream>>>(sc, src_mask, SSK, NH, STQ,
                                                     (long long)SSK, 0LL);
        launch_gemm(stream, sc, kv + HD, nullptr, nullptr, attn,
                    STQ, HDD, SSK, SSK, 2 * HD, HD,
                    (long long)NH * STQ * SSK, (long long)STQ * SSK,
                    (long long)SSK * 2 * HD, HDD,
                    (long long)STQ * HD, HDD,
                    NH, BH, 1.f, 1 | 4);
        launch_wt(stream, cow, wt, HD, TDIM);
        launch_gemm(stream, attn, wt, cob, t, t,
                    Mq, TDIM, HD, HD, HD, TDIM,
                    0,0,0,0,0,0, 1, 1, 1.f, 0);

        // ===================== feed-forward =====================
        ln_kernel<<<Mq, 256, 0, stream>>>(t, pg, pb, (unsigned int*)tn, TDIM);
        launch_wt(stream, w1, wt, TDIM, FFD);
        launch_gemm(stream, tn, wt, b1, nullptr, sc,            // sc reused as FF hidden
                    Mq, FFD, TDIM, TDIM, TDIM, FFD,
                    0,0,0,0,0,0, 1, 1, 1.f, 2 | 4);
        launch_wt(stream, w2, wt, FFD, TDIM);
        launch_gemm(stream, sc, wt, b2, t, t,
                    Mq, TDIM, FFD, FFD, FFD, TDIM,
                    0,0,0,0,0,0, 1, 1, 1.f, 0);
    }
    (void)in_sizes; (void)n_in; (void)out_size; (void)ws_size;
}
